// RNN_54374285968098
// MI455X (gfx1250) — matrix operand order repaired and checked by reading, not compiled, not run on hardware
//
#include <hip/hip_runtime.h>

// ---------------------------------------------------------------------------
// GRU on MI455X (gfx1250): persistent-kernel, LDS-resident bf16 weights,
// v_wmma_f32_16x16x32_bf16 inner loop, atomic grid barriers per timestep.
// ---------------------------------------------------------------------------

typedef __bf16 bf16;
typedef __attribute__((ext_vector_type(16))) __bf16 v16bf;
typedef __attribute__((ext_vector_type(8)))  float  v8f;

#define B_    32
#define T_    512
#define F_    1024
#define U_    1024
#define K2_   2048          // F_ + U_
#define NWG   64            // one workgroup per WGP
#define TPB   128           // 4 wave32
#define GCOLS 32            // gate columns per WG (2048 / 64)
#define CCOLS 16            // candidate columns per WG (1024 / 64)
#define KP    2056          // padded LDS column stride (bf16) -> conflict-free halves
#define GATE_LDS_BYTES (GCOLS * KP * 2)          // 131584
#define RED_OFF        GATE_LDS_BYTES
#define LDS_BYTES      (GATE_LDS_BYTES + 2*256*4) // + K-split reduction buffer

union ABfrag { uint4 q[2]; v16bf v; };

// Fast activations: hardware v_rcp_f32 instead of IEEE div fixup chains.
__device__ inline float fast_rcp(float x) { return __builtin_amdgcn_rcpf(x); }
__device__ inline float fast_sigmoid(float x) {
    return fast_rcp(1.0f + __expf(-x));
}
__device__ inline float fast_tanh(float x) {
    x = fminf(15.0f, fmaxf(-15.0f, x));
    float e = __expf(2.0f * x);
    return (e - 1.0f) * fast_rcp(e + 1.0f);
}

// Device-scope grid barrier: monotonic counter, thread 0 arrives+spins.
__device__ inline void grid_sync(unsigned* bar, unsigned target) {
    __threadfence();          // release: make hbuf/ubuf/rhbf/hbf visible at L2
    __syncthreads();
    if (threadIdx.x == 0) {
        atomicAdd(bar, 1u);
        while (__hip_atomic_load(bar, __ATOMIC_RELAXED, __HIP_MEMORY_SCOPE_AGENT) < target)
            __builtin_amdgcn_s_sleep(2);
    }
    __syncthreads();
    __threadfence();          // acquire: invalidate stale L0 lines
}

// ------------------------- prep kernels ------------------------------------

// inputs [B,T,F] f32  ->  xbf [T,B,F] bf16
__global__ void k_convert_x(const float* __restrict__ x, bf16* __restrict__ xbf) {
    const long long n = (long long)T_ * B_ * F_;
    for (long long i = (long long)blockIdx.x * blockDim.x + threadIdx.x; i < n;
         i += (long long)gridDim.x * blockDim.x) {
        long long t = i / ((long long)B_ * F_);
        long long r = i % ((long long)B_ * F_);
        long long b = r / F_, f = r % F_;
        xbf[i] = (bf16)x[(b * T_ + t) * F_ + f];
    }
}

// w [K,N] f32 row-major  ->  wT [N,K] bf16 (K contiguous per column)
__global__ void k_transpose_w(const float* __restrict__ w, bf16* __restrict__ wT,
                              int K, int N) {
    const long long n = (long long)K * N;
    for (long long i = (long long)blockIdx.x * blockDim.x + threadIdx.x; i < n;
         i += (long long)gridDim.x * blockDim.x) {
        long long col = i / K, k = i % K;
        wT[i] = (bf16)w[k * (long long)N + col];
    }
}

__global__ void k_init(float* __restrict__ hbuf, bf16* __restrict__ hbf,
                       unsigned* __restrict__ bar) {
    for (int i = blockIdx.x * blockDim.x + threadIdx.x; i < B_ * U_;
         i += gridDim.x * blockDim.x) {
        hbuf[i] = 0.0f;
        hbf[i]  = (bf16)0.0f;
    }
    if (blockIdx.x == 0 && threadIdx.x < 16) bar[threadIdx.x] = 0u;
}

// ------------------------- persistent GRU kernel ---------------------------

__global__ __launch_bounds__(TPB, 1)
void gru_persistent(const bf16* __restrict__ xbf,   // [T,B,F] bf16
                    const bf16* __restrict__ gateT, // [2048 cols][2048 K] bf16
                    const bf16* __restrict__ candT, // [1024 cols][2048 K] bf16
                    const float* __restrict__ gate_bias,
                    const float* __restrict__ cand_bias,
                    float* __restrict__ hbuf,       // [B,U] f32
                    bf16*  __restrict__ hbf,        // [B,U] bf16
                    bf16*  __restrict__ rhbf,       // [B,U] bf16 (r*h)
                    float* __restrict__ ubuf,       // [B,U] f32 (update gate)
                    float* __restrict__ out,        // [B,T,U] f32
                    unsigned* __restrict__ bar) {
    extern __shared__ char smem[];
    bf16*  wlds = (bf16*)smem;                 // gate weight slice [GCOLS][KP]
    float* red  = (float*)(smem + RED_OFF);    // cand K-split reduction

    const int wg   = blockIdx.x;       // 0..63
    const int tid  = threadIdx.x;
    const int lane = tid & 31;
    const int wave = tid >> 5;         // 0..3
    const int half = lane >> 4;        // 0|1: K-half of fragment per WMMA layout
    const int ln   = lane & 15;
    const int hoff = half ? 8 : 0;     // A-fragment lane K offset (elements)

    // ---- stage this WG's gate weight slice into LDS (resident for all T) ----
    for (int idx = tid; idx < GCOLS * (K2_ / 8); idx += TPB) {
        int col = idx / (K2_ / 8);
        int k8  = (idx % (K2_ / 8)) * 8;
        uint4 v = *(const uint4*)(gateT + ((size_t)(wg * GCOLS + col)) * K2_ + k8);
        *(uint4*)(wlds + (size_t)col * KP + k8) = v;
    }
    __syncthreads();

    // phase-A tile for this wave: 2 M-tiles x 2 N-tiles
    const int mtile = wave & 1;
    const int ntile = wave >> 1;
    const int mA    = mtile * 16 + ln;                 // A row (batch)
    const int gc    = wg * GCOLS + ntile * 16 + ln;    // gate column
    const float gbias = gate_bias[gc];
    const bf16* bbaseA = wlds + (size_t)(ntile * 16 + ln) * KP + half * 8;

    // phase-B: K split across wave pairs; waves 0/1 own output tiles
    const int mB = (wave & 1) * 16 + ln;
    const int kh = wave >> 1;                          // 0: K[0,1024) 1: K[1024,2048)
    const int nC = wg * CCOLS + ln;                    // candidate column
    const float cbias = cand_bias[nC];
    const bf16* bbaseB = candT + (size_t)nC * K2_ + kh * F_ + half * 8;

    unsigned target = 0;

    for (int t = 0; t < T_; ++t) {
        // ================= phase A: gates = sigmoid([x|h] @ Wg + bg) ========
        v8f acc = {};
        {
            // K in [0, 1024): A from x_t;  K in [1024, 2048): A from h_{t-1}
            const bf16* ap = xbf + ((size_t)t * B_ + mA) * F_ + hoff;
            const bf16* bp = bbaseA;
            #pragma unroll 4
            for (int kb = 0; kb < F_ / 32; ++kb) {
                __builtin_prefetch(ap + 64, 0, 1);     // global_prefetch_b8
                ABfrag a, b;
                a.q[0] = *(const uint4*)(ap);
                a.q[1] = *(const uint4*)(ap + 16);
                b.q[0] = *(const uint4*)(bp);
                b.q[1] = *(const uint4*)(bp + 16);
                acc = __builtin_amdgcn_wmma_f32_16x16x32_bf16(
                    false, a.v, false, b.v, (short)0, acc, false, false);
                ap += 32;
                bp += 32;
            }
            ap = hbf + (size_t)mA * U_ + hoff;
            #pragma unroll 4
            for (int kb = 0; kb < U_ / 32; ++kb) {
                __builtin_prefetch(ap + 64, 0, 1);
                ABfrag a, b;
                a.q[0] = *(const uint4*)(ap);
                a.q[1] = *(const uint4*)(ap + 16);
                b.q[0] = *(const uint4*)(bp);
                b.q[1] = *(const uint4*)(bp + 16);
                acc = __builtin_amdgcn_wmma_f32_16x16x32_bf16(
                    false, a.v, false, b.v, (short)0, acc, false, false);
                ap += 32;
                bp += 32;
            }
        }
        #pragma unroll
        for (int r = 0; r < 8; ++r) {
            const int br = mtile * 16 + half * 8 + r;  // batch row of element r
            const float g = fast_sigmoid(acc[r] + gbias);
            if (wg < 32) {   // reset gate -> stage r*h in bf16 for cand GEMM
                const size_t hi = (size_t)br * U_ + gc;
                rhbf[hi] = (bf16)(g * hbuf[hi]);
            } else {         // update gate
                ubuf[(size_t)br * U_ + (gc - U_)] = g;
            }
        }
        target += NWG;
        grid_sync(bar, target);

        // ============ phase B: c = tanh([x|r*h] @ Wc + bc); h update ========
        // This wave's K-half lies entirely in one A source: hoist the select.
        v8f acc2 = {};
        {
            const bf16* ap = (kh == 0 ? xbf + ((size_t)t * B_ + mB) * F_
                                      : rhbf + (size_t)mB * U_) + hoff;
            const bf16* bp = bbaseB;
            #pragma unroll 4
            for (int kb = 0; kb < F_ / 32; ++kb) {
                __builtin_prefetch(ap + 64, 0, 1);
                ABfrag a, b;
                a.q[0] = *(const uint4*)(ap);
                a.q[1] = *(const uint4*)(ap + 16);
                b.q[0] = *(const uint4*)(bp);
                b.q[1] = *(const uint4*)(bp + 16);
                acc2 = __builtin_amdgcn_wmma_f32_16x16x32_bf16(
                    false, a.v, false, b.v, (short)0, acc2, false, false);
                ap += 32;
                bp += 32;
            }
        }
        if (wave >= 2) {     // store K-upper partial
            #pragma unroll
            for (int r = 0; r < 8; ++r)
                red[(wave - 2) * 256 + lane * 8 + r] = acc2[r];
        }
        __syncthreads();
        if (wave < 2) {      // reduce, activate, update h, emit output
            #pragma unroll
            for (int r = 0; r < 8; ++r) {
                float c = acc2[r] + red[wave * 256 + lane * 8 + r] + cbias;
                c = fast_tanh(c);
                const int br = (wave & 1) * 16 + half * 8 + r;
                const size_t hi = (size_t)br * U_ + nC;
                const float u  = ubuf[hi];
                const float hn = u * hbuf[hi] + (1.0f - u) * c;
                out[((size_t)br * T_ + t) * U_ + nC] = hn;
                hbuf[hi] = hn;
                hbf[hi]  = (bf16)hn;
            }
        }
        target += NWG;
        grid_sync(bar, target);
    }
}

// ------------------------- launch ------------------------------------------

extern "C" void kernel_launch(void* const* d_in, const int* in_sizes, int n_in,
                              void* d_out, int out_size, void* d_ws, size_t ws_size,
                              hipStream_t stream) {
    const float* x  = (const float*)d_in[0];  // [B,T,F]
    const float* gk = (const float*)d_in[1];  // [2048,2048]
    const float* gb = (const float*)d_in[2];  // [2048]
    const float* ck = (const float*)d_in[3];  // [2048,1024]
    const float* cb = (const float*)d_in[4];  // [1024]
    float* out = (float*)d_out;

    char* ws = (char*)d_ws;
    bf16*     xbf   = (bf16*)(ws);                        // 32 MB
    bf16*     gateT = (bf16*)(ws + 33554432);             //  8 MB
    bf16*     candT = (bf16*)(ws + 41943040);             //  4 MB
    float*    hbuf  = (float*)(ws + 46137344);            // 128 KB
    float*    ubuf  = (float*)(ws + 46268416);            // 128 KB
    bf16*     hbf   = (bf16*)(ws + 46399488);             //  64 KB
    bf16*     rhbf  = (bf16*)(ws + 46465024);             //  64 KB
    unsigned* bar   = (unsigned*)(ws + 46530560);

    (void)in_sizes; (void)n_in; (void)out_size; (void)ws_size;

    hipFuncSetAttribute(reinterpret_cast<const void*>(&gru_persistent),
                        hipFuncAttributeMaxDynamicSharedMemorySize, LDS_BYTES);

    k_convert_x <<<4096, 256, 0, stream>>>(x, xbf);
    k_transpose_w<<<4096, 256, 0, stream>>>(gk, gateT, K2_, 2048);
    k_transpose_w<<<2048, 256, 0, stream>>>(ck, candT, K2_, 1024);
    k_init      <<<64,   256, 0, stream>>>(hbuf, hbf, bar);
    gru_persistent<<<NWG, TPB, LDS_BYTES, stream>>>(
        xbf, gateT, candT, gb, cb, hbuf, hbf, rhbf, ubuf, out, bar);
}
